// SpatialPooler_6992206758563
// MI455X (gfx1250) — 8-bit matrix operand order repaired and checked by reading, compiled, not run on hardware
//
#include <hip/hip_runtime.h>
#include <hip/hip_bf16.h>
#include <math.h>
#include <stdint.h>

#define BATCH   1024
#define IN_DIM  4096
#define OUT_DIM 8192
#define TOP_K   164
#define BOOST_STRENGTH 100.0f

typedef __attribute__((ext_vector_type(8))) int v8i;

union FragA { v8i v; unsigned long long d[4]; };
union FragB { v8i v; uint4 q[2]; };

__global__ __launch_bounds__(256) void quant_kernel(const float* __restrict__ in,
                                                    unsigned char* __restrict__ out,
                                                    int n) {
    int i = (blockIdx.x * 256 + threadIdx.x) * 4;
    if (i + 3 < n) {
        float4 v = *(const float4*)(in + i);
        unsigned w = (unsigned)(v.x != 0.0f)
                   | ((unsigned)(v.y != 0.0f) << 8)
                   | ((unsigned)(v.z != 0.0f) << 16)
                   | ((unsigned)(v.w != 0.0f) << 24);
        *(unsigned*)(out + i) = w;
    }
}

__global__ __launch_bounds__(256) void quantT_kernel(const float* __restrict__ in,
                                                     unsigned char* __restrict__ outT) {
    __shared__ unsigned char tile[64][80];
    const int n0 = blockIdx.x * 64;
    const int k0 = blockIdx.y * 64;
    const int t  = threadIdx.x;

    const int tn = (t & 15) * 4;
    const int tk = t >> 4;
    #pragma unroll
    for (int i = 0; i < 4; ++i) {
        int kk = tk + 16 * i;
        float4 v = *(const float4*)(in + (size_t)(k0 + kk) * OUT_DIM + n0 + tn);
        tile[tn + 0][kk] = (unsigned char)(v.x != 0.0f);
        tile[tn + 1][kk] = (unsigned char)(v.y != 0.0f);
        tile[tn + 2][kk] = (unsigned char)(v.z != 0.0f);
        tile[tn + 3][kk] = (unsigned char)(v.w != 0.0f);
    }
    __syncthreads();

    const int wn = t >> 2;
    const int wk = (t & 3) * 16;
    uint4 q = *(const uint4*)(&tile[wn][wk]);
    *(uint4*)(outT + (size_t)(n0 + wn) * IN_DIM + k0 + wk) = q;
}

__global__ __launch_bounds__(256) void boost_kernel(const float* __restrict__ avg,
                                                    float* __restrict__ boost) {
    __shared__ float red[256];
    int t = threadIdx.x;
    float s = 0.0f;
    for (int j = t; j < OUT_DIM; j += 256) s += avg[j];
    red[t] = s;
    __syncthreads();
    for (int o = 128; o > 0; o >>= 1) {
        if (t < o) red[t] += red[t + o];
        __syncthreads();
    }
    float S = red[0];
    const float inv = 1.0f / (float)(OUT_DIM - 1);
    for (int j = t; j < OUT_DIM; j += 256) {
        float a = avg[j];
        float neigh = (S - a) * inv;
        boost[j] = __expf(-BOOST_STRENGTH * (a - neigh));
    }
}

#define TILE_M 128
#define TILE_N 128
#define TILE_K 64
#define LDSP   80

__global__ __launch_bounds__(256) void gemm_iu8_kernel(
        const unsigned char* __restrict__ Aq,
        const unsigned char* __restrict__ BqT,
        const float* __restrict__ boost,
        float* __restrict__ overlap)
{
    __shared__ unsigned char Ash[2][TILE_M * LDSP];
    __shared__ unsigned char Bsh[2][TILE_N * LDSP];

    const int t     = threadIdx.x;
    const int lane  = t & 31;
    const int wave  = t >> 5;
    const int waveM = wave >> 2;
    const int waveN = wave & 3;
    const int blockN = blockIdx.x * TILE_N;
    const int blockM = blockIdx.y * TILE_M;

    v8i acc[4][2];
    const v8i vzero = {0, 0, 0, 0, 0, 0, 0, 0};
    for (int mt = 0; mt < 4; ++mt)
        for (int nt = 0; nt < 2; ++nt)
            acc[mt][nt] = vzero;

    const int s_r  = t >> 1;
    const int s_kc = (t & 1) * 32;
    const unsigned char* gA = Aq  + (size_t)(blockM + s_r) * IN_DIM + s_kc;
    const unsigned char* gB = BqT + (size_t)(blockN + s_r) * IN_DIM + s_kc;
    const unsigned lA0 = (unsigned)(uintptr_t)&Ash[0][s_r * LDSP + s_kc];
    const unsigned lB0 = (unsigned)(uintptr_t)&Bsh[0][s_r * LDSP + s_kc];
    const unsigned lA1 = (unsigned)(uintptr_t)&Ash[1][s_r * LDSP + s_kc];
    const unsigned lB1 = (unsigned)(uintptr_t)&Bsh[1][s_r * LDSP + s_kc];

    #define STAGE(buf, koff)                                                         \
        do {                                                                         \
            unsigned la = (buf) ? lA1 : lA0;                                         \
            unsigned lb = (buf) ? lB1 : lB0;                                         \
            unsigned long long ga = (unsigned long long)(uintptr_t)(gA + (koff));    \
            unsigned long long gb = (unsigned long long)(uintptr_t)(gB + (koff));    \
            asm volatile(                                                            \
                "global_load_async_to_lds_b128 %0, %2, off\n\t"                      \
                "global_load_async_to_lds_b128 %0, %2, off offset:16\n\t"            \
                "global_load_async_to_lds_b128 %1, %3, off\n\t"                      \
                "global_load_async_to_lds_b128 %1, %3, off offset:16"                \
                :: "v"(la), "v"(lb), "v"(ga), "v"(gb) : "memory");                   \
        } while (0)

    const int klo  = (lane & 16) ? 8  : 0;
    const int klo2 = (lane & 16) ? 16 : 0;
    const int lm   = lane & 15;

    STAGE(0, 0);

    for (int k0 = 0; k0 < IN_DIM; k0 += TILE_K) {
        const int buf = (k0 >> 6) & 1;

        asm volatile("s_wait_asynccnt 0x0" ::: "memory");
        __syncthreads();

        if (k0 + TILE_K < IN_DIM) {
            STAGE(buf ^ 1, k0 + TILE_K);
        }

        const unsigned char* As = &Ash[buf][0];
        const unsigned char* Bs = &Bsh[buf][0];

        FragB bf[2];
        #pragma unroll
        for (int nt = 0; nt < 2; ++nt) {
            int n = waveN * 32 + nt * 16 + lm;
            bf[nt].q[0] = *(const uint4*)(&Bs[n * LDSP + klo2     ]);
            bf[nt].q[1] = *(const uint4*)(&Bs[n * LDSP + klo2 + 32]);
        }
        #pragma unroll
        for (int mt = 0; mt < 4; ++mt) {
            int m = waveM * 64 + mt * 16 + lm;
            FragA af;
            af.d[0] = *(const unsigned long long*)(&As[m * LDSP + klo2     ]);
            af.d[1] = *(const unsigned long long*)(&As[m * LDSP + klo2 +  8]);
            af.d[2] = *(const unsigned long long*)(&As[m * LDSP + klo2 + 32]);
            af.d[3] = *(const unsigned long long*)(&As[m * LDSP + klo2 + 40]);
            #pragma unroll
            for (int nt = 0; nt < 2; ++nt) {
                acc[mt][nt] = __builtin_amdgcn_wmma_i32_16x16x64_iu8(
                     false, af.v,  false, bf[nt].v,
                    acc[mt][nt],  false,  false);
            }
        }
    }

    const int rbase = (lane < 16) ? 0 : 8;
    #pragma unroll
    for (int nt = 0; nt < 2; ++nt) {
        int ccol = blockN + waveN * 32 + nt * 16 + lm;
        float bst = boost[ccol];
        #pragma unroll
        for (int mt = 0; mt < 4; ++mt) {
            int rowbase = blockM + waveM * 64 + mt * 16 + rbase;
            #pragma unroll
            for (int r = 0; r < 8; ++r) {
                overlap[(size_t)(rowbase + r) * OUT_DIM + ccol] =
                    (float)acc[mt][nt][r] * bst;
            }
        }
    }
    #undef STAGE
}

__global__ __launch_bounds__(256) void topk_kernel(const float* __restrict__ overlap,
                                                   float* __restrict__ out) {
    __shared__ unsigned keys[OUT_DIM];
    __shared__ int cnt;
    __shared__ int tie;

    const int row = blockIdx.x;
    const int t   = threadIdx.x;
    const unsigned* src = (const unsigned*)(overlap + (size_t)row * OUT_DIM);

    for (int j = t; j < OUT_DIM; j += 256) keys[j] = src[j];
    __syncthreads();

    unsigned T = 0u;
    for (int bit = 31; bit >= 0; --bit) {
        unsigned cand = T | (1u << bit);
        if (t == 0) cnt = 0;
        __syncthreads();
        int c = 0;
        for (int j = t; j < OUT_DIM; j += 256) c += (keys[j] >= cand);
        atomicAdd(&cnt, c);
        __syncthreads();
        if (cnt >= TOP_K) T = cand;
        __syncthreads();
    }

    if (t == 0) { cnt = 0; tie = 0; }
    __syncthreads();
    int c = 0;
    for (int j = t; j < OUT_DIM; j += 256) c += (keys[j] > T);
    atomicAdd(&cnt, c);
    __syncthreads();
    const int quota = TOP_K - cnt;

    float* dst = out + (size_t)row * OUT_DIM;
    for (int j = t; j < OUT_DIM; j += 256) {
        unsigned k = keys[j];
        float v;
        if (k > T) {
            v = 1.0f;
        } else if (k == T) {
            int p = atomicAdd(&tie, 1);
            v = (p < quota) ? 1.0f : 0.0f;
        } else {
            v = 0.0f;
        }
        dst[j] = v;
    }
}

extern "C" void kernel_launch(void* const* d_in, const int* in_sizes, int n_in,
                              void* d_out, int out_size, void* d_ws, size_t ws_size,
                              hipStream_t stream) {
    (void)in_sizes; (void)n_in; (void)out_size; (void)ws_size;

    const float* x    = (const float*)d_in[0];
    const float* conn = (const float*)d_in[1];
    const float* avg  = (const float*)d_in[2];
    float* out = (float*)d_out;

    char* ws = (char*)d_ws;
    float*         overlap = (float*)ws;
    float*         boost   = (float*)(ws + 33554432);
    unsigned char* xq      = (unsigned char*)(ws + 33554432 + 32768);
    unsigned char* connqT  = (unsigned char*)(ws + 33554432 + 32768 + 4194304);

    const int nx = BATCH * IN_DIM;

    quant_kernel<<<nx / 1024, 256, 0, stream>>>(x, xq, nx);
    dim3 tgrid(OUT_DIM / 64, IN_DIM / 64);
    quantT_kernel<<<tgrid, 256, 0, stream>>>(conn, connqT);
    boost_kernel<<<1, 256, 0, stream>>>(avg, boost);

    dim3 ggrid(OUT_DIM / TILE_N, BATCH / TILE_M);
    gemm_iu8_kernel<<<ggrid, 256, 0, stream>>>(xq, connqT, boost, overlap);

    topk_kernel<<<BATCH, 256, 0, stream>>>(overlap, out);
}
